// MoELayer_81535659147915
// MI455X (gfx1250) — matrix operand order repaired and checked by reading, not compiled, not run on hardware
//
#include <hip/hip_runtime.h>

#define Bb 4
#define Tt 2048
#define Dd 1024
#define Ee 8
#define Hh 2048
#define Nn (Bb*Tt)     // 8192 tokens
#define CAP 8192       // max tokens per expert (top-2 distinct => <= N)

typedef __attribute__((ext_vector_type(16))) __bf16 v16bf;
typedef __attribute__((ext_vector_type(8)))  __bf16 v8bf;
typedef __attribute__((ext_vector_type(8)))  float  v8f;
typedef __attribute__((ext_vector_type(4)))  float  v4f;
typedef __attribute__((ext_vector_type(4)))  int    v4i;

#if defined(__has_builtin)
#if __has_builtin(__builtin_amdgcn_global_load_async_to_lds_b128) && \
    __has_builtin(__builtin_amdgcn_s_wait_asynccnt)
#define USE_ASYNC_LDS 1
#endif
#endif

// ws layout (bytes)
#define WS_COUNTS   0
#define WS_LIST     1024
#define WS_GATES    (1024 + Ee * CAP * 4)
#define WS_HBUF     (512 * 1024)
#define WS_W1T      67633152ull            // after hbuf (64 MB), 256-aligned
#define WS_W2T      101187584ull
#define WS_NEED     134742016ull

// ---------------- zero kernel: out + expert counters ----------------
__global__ __launch_bounds__(256) void moe_zero(float* __restrict__ out, int n,
                                                int* __restrict__ counts) {
    if (blockIdx.x == 0 && threadIdx.x < Ee) counts[threadIdx.x] = 0;
    for (int i = blockIdx.x * blockDim.x + threadIdx.x; i < n;
         i += gridDim.x * blockDim.x)
        out[i] = 0.0f;
}

// ------------- weight transpose + bf16 convert: in[R][C] -> outT[C][R] -------------
__global__ __launch_bounds__(256) void moe_transpose(const float* __restrict__ in,
                                                     __bf16* __restrict__ outT,
                                                     int R, int C) {
    __shared__ float tile[32][33];
    const size_t plane = (size_t)R * C;
    const float* src = in + (size_t)blockIdx.z * plane;
    __bf16* dst = outT + (size_t)blockIdx.z * plane;
    const int c0 = blockIdx.x * 32, r0 = blockIdx.y * 32;
    const int tx = threadIdx.x & 31, ty = threadIdx.x >> 5;   // ty: 0..7
#pragma unroll
    for (int k = 0; k < 4; ++k)
        tile[ty + 8 * k][tx] = src[(size_t)(r0 + ty + 8 * k) * C + (c0 + tx)];
    __syncthreads();
#pragma unroll
    for (int k = 0; k < 4; ++k)
        dst[(size_t)(c0 + ty + 8 * k) * R + (r0 + tx)] = (__bf16)tile[tx][ty + 8 * k];
}

// ---------------- router: logits, top-2 softmax, scatter ----------------
__global__ __launch_bounds__(256) void moe_router(const float* __restrict__ x,
                                                  const float* __restrict__ rw,
                                                  int* __restrict__ counts,
                                                  int* __restrict__ list,
                                                  float* __restrict__ gates) {
    __shared__ float s_rw[Ee * Dd];          // 32 KB of 320 KB LDS
    const int tid = threadIdx.x;
    for (int i = tid; i < Ee * Dd; i += 256) s_rw[i] = rw[i];
    __syncthreads();

    const int wave = tid >> 5, lane = tid & 31;
    const int n = blockIdx.x * 8 + wave;
    const float* xr = x + (size_t)n * Dd;

    float acc[Ee];
#pragma unroll
    for (int e = 0; e < Ee; ++e) acc[e] = 0.0f;
    for (int d = lane; d < Dd; d += 32) {
        float xv = xr[d];
#pragma unroll
        for (int e = 0; e < Ee; ++e) acc[e] += xv * s_rw[e * Dd + d];
    }
#pragma unroll
    for (int e = 0; e < Ee; ++e) {
#pragma unroll
        for (int off = 16; off > 0; off >>= 1)
            acc[e] += __shfl_xor(acc[e], off, 32);
    }
    if (lane == 0) {
        float best = -1e30f; int be = 0;
#pragma unroll
        for (int e = 0; e < Ee; ++e)
            if (acc[e] > best) { best = acc[e]; be = e; }
        float sec = -1e30f; int se = (be == 0) ? 1 : 0;
#pragma unroll
        for (int e = 0; e < Ee; ++e)
            if (e != be && acc[e] > sec) { sec = acc[e]; se = e; }
        float p1 = __expf(sec - best);
        float inv = 1.0f / (1.0f + p1);
        int pos0 = atomicAdd(&counts[be], 1);
        list[be * CAP + pos0] = n * 2;
        gates[n * 2] = inv;
        int pos1 = atomicAdd(&counts[se], 1);
        list[se * CAP + pos1] = n * 2 + 1;
        gates[n * 2 + 1] = p1 * inv;
    }
}

// ---------------- FFN1: h = gate * relu(x @ w1[e])^2  (bf16 out) ----------------
// Block: 128 thr (4 waves). Tile: M=64 x N=128. A staged in LDS (bf16).
// B: contiguous b128 loads from pre-transposed bf16 w1t (use_tr) or strided
// fp32 gather fallback. Per wave: 2 B frags, 8 accumulators, 8 WMMA / k-step.
__global__ __launch_bounds__(128) void moe_ffn1(const float* __restrict__ x,
                                                const float* __restrict__ w1,
                                                const __bf16* __restrict__ w1t,
                                                int use_tr,
                                                const int* __restrict__ counts,
                                                const int* __restrict__ list,
                                                const float* __restrict__ gates,
                                                __bf16* __restrict__ hbuf) {
    __shared__ __bf16 sA[64 * 32];           // 4 KB staging tile
    const int e = blockIdx.z;
    const int cnt = counts[e];
    const int m0 = blockIdx.x * 64;
    if (m0 >= cnt) return;
    const int tid = threadIdx.x;
    const int wave = tid >> 5, lane = tid & 31;
    const int lh = lane & 15, hi = lane >> 4;
    const int* elist = list + e * CAP;

    const int sr = tid >> 1, shalf = tid & 1;
    const int mstage = (m0 + sr) < cnt ? (m0 + sr) : cnt - 1;
    const float* stage_src = x + (size_t)(elist[mstage] >> 1) * Dd + shalf * 16;
    __bf16* stage_dst = &sA[sr * 32 + shalf * 16];

    const int nbase = blockIdx.y * 128 + wave * 32;
    const float* bp0 = w1 + (size_t)e * Dd * Hh + (nbase + lh);
    const float* bp1 = bp0 + 16;
    const __bf16* bt0 = w1t + (size_t)e * Dd * Hh + (size_t)(nbase + lh) * Dd;
    const __bf16* bt1 = bt0 + (size_t)16 * Dd;

    const v8f vzero = {0.f, 0.f, 0.f, 0.f, 0.f, 0.f, 0.f, 0.f};
    v8f acc[4][2];
#pragma unroll
    for (int mf = 0; mf < 4; ++mf) { acc[mf][0] = vzero; acc[mf][1] = vzero; }

    for (int k0 = 0; k0 < Dd; k0 += 32) {
        __syncthreads();                     // protect previous round's reads
        {   // stage A tile (fp32 -> bf16), 16 floats per thread
            const float* ap = stage_src + k0;
            __builtin_prefetch(ap + 32, 0, 1);
            v4f a0 = *(const v4f*)(ap);
            v4f a1 = *(const v4f*)(ap + 4);
            v4f a2 = *(const v4f*)(ap + 8);
            v4f a3 = *(const v4f*)(ap + 12);
            v8bf lo, hs;
#pragma unroll
            for (int j = 0; j < 4; ++j) {
                lo[j]     = (__bf16)a0[j];
                lo[4 + j] = (__bf16)a1[j];
                hs[j]     = (__bf16)a2[j];
                hs[4 + j] = (__bf16)a3[j];
            }
            *(v8bf*)(stage_dst)     = lo;
            *(v8bf*)(stage_dst + 8) = hs;
        }
        // B fragments
        v16bf bfrag0, bfrag1;
        const int kb = k0 + hi * 16;
        if (use_tr) {                        // contiguous bf16 (transposed weights)
            v8bf l0 = *(const v8bf*)(bt0 + kb);
            v8bf h0 = *(const v8bf*)(bt0 + kb + 8);
            v8bf l1 = *(const v8bf*)(bt1 + kb);
            v8bf h1 = *(const v8bf*)(bt1 + kb + 8);
#pragma unroll
            for (int j = 0; j < 8; ++j) {
                bfrag0[j] = l0[j]; bfrag0[8 + j] = h0[j];
                bfrag1[j] = l1[j]; bfrag1[8 + j] = h1[j];
            }
        } else {                             // strided fp32 gather fallback
#pragma unroll
            for (int j = 0; j < 16; ++j) {
                bfrag0[j] = (__bf16)bp0[(size_t)(kb + j) * Hh];
                bfrag1[j] = (__bf16)bp1[(size_t)(kb + j) * Hh];
            }
        }
        __syncthreads();                     // staging visible
#pragma unroll
        for (int mf = 0; mf < 4; ++mf) {
            const __bf16* ar = &sA[(mf * 16 + lh) * 32 + hi * 16];
            v8bf lo = *(const v8bf*)(ar);        // K {0..7}+hi*16
            v8bf hs = *(const v8bf*)(ar + 8);    // K {8..15}+hi*16
            v16bf afrag;
#pragma unroll
            for (int j = 0; j < 8; ++j) { afrag[j] = lo[j]; afrag[8 + j] = hs[j]; }
            acc[mf][0] = __builtin_amdgcn_wmma_f32_16x16x32_bf16(
                false, afrag, false, bfrag0, (short)0, acc[mf][0], false, false);
            acc[mf][1] = __builtin_amdgcn_wmma_f32_16x16x32_bf16(
                false, afrag, false, bfrag1, (short)0, acc[mf][1], false, false);
        }
    }
#pragma unroll
    for (int mf = 0; mf < 4; ++mf) {
#pragma unroll
        for (int i = 0; i < 8; ++i) {
            int m = m0 + mf * 16 + i + hi * 8;
            if (m < cnt) {
                int id = elist[m];
                float g = gates[id];
                __bf16* hrow = hbuf + (size_t)id * Hh;
#pragma unroll
                for (int nf = 0; nf < 2; ++nf) {
                    float v = acc[mf][nf][i];
                    v = v > 0.0f ? v : 0.0f;
                    v = v * v * g;
                    hrow[nbase + nf * 16 + lh] = (__bf16)v;
                }
            }
        }
    }
}

// ---------------- FFN2: out += h @ w2[e]  (two commutative atomics/token) ------
__global__ __launch_bounds__(128) void moe_ffn2(const __bf16* __restrict__ hbuf,
                                                const float* __restrict__ w2,
                                                const __bf16* __restrict__ w2t,
                                                int use_tr,
                                                const int* __restrict__ counts,
                                                const int* __restrict__ list,
                                                float* __restrict__ out) {
    __shared__ __bf16 sA[64 * 32];           // 4 KB staging tile
    const int e = blockIdx.z;
    const int cnt = counts[e];
    const int m0 = blockIdx.x * 64;
    if (m0 >= cnt) return;
    const int tid = threadIdx.x;
    const int wave = tid >> 5, lane = tid & 31;
    const int lh = lane & 15, hi = lane >> 4;
    const int* elist = list + e * CAP;

    const int sr = tid >> 1, shalf = tid & 1;
    const int mstage = (m0 + sr) < cnt ? (m0 + sr) : cnt - 1;
    const __bf16* stage_src = hbuf + (size_t)elist[mstage] * Hh + shalf * 16;
    __bf16* stage_dst = &sA[sr * 32 + shalf * 16];

    const int nbase = blockIdx.y * 128 + wave * 32;
    const float* bp0 = w2 + (size_t)e * Hh * Dd + (nbase + lh);
    const float* bp1 = bp0 + 16;
    const __bf16* bt0 = w2t + (size_t)e * Hh * Dd + (size_t)(nbase + lh) * Hh;
    const __bf16* bt1 = bt0 + (size_t)16 * Hh;

    const v8f vzero = {0.f, 0.f, 0.f, 0.f, 0.f, 0.f, 0.f, 0.f};
    v8f acc[4][2];
#pragma unroll
    for (int mf = 0; mf < 4; ++mf) { acc[mf][0] = vzero; acc[mf][1] = vzero; }

    for (int k0 = 0; k0 < Hh; k0 += 32) {
        __syncthreads();
        {   // stage A tile: raw bf16 copy, 32 B per thread (async global->LDS)
            const __bf16* ap = stage_src + k0;
#if defined(USE_ASYNC_LDS)
            __builtin_amdgcn_global_load_async_to_lds_b128(
                (v4i*)(ap), (v4i*)(stage_dst), 0, 0);
            __builtin_amdgcn_global_load_async_to_lds_b128(
                (v4i*)(ap + 8), (v4i*)(stage_dst + 8), 0, 0);
#else
            v8bf lo = *(const v8bf*)(ap);
            v8bf hs = *(const v8bf*)(ap + 8);
            *(v8bf*)(stage_dst)     = lo;
            *(v8bf*)(stage_dst + 8) = hs;
#endif
        }
        v16bf bfrag0, bfrag1;
        const int kb = k0 + hi * 16;
        if (use_tr) {
            v8bf l0 = *(const v8bf*)(bt0 + kb);
            v8bf h0 = *(const v8bf*)(bt0 + kb + 8);
            v8bf l1 = *(const v8bf*)(bt1 + kb);
            v8bf h1 = *(const v8bf*)(bt1 + kb + 8);
#pragma unroll
            for (int j = 0; j < 8; ++j) {
                bfrag0[j] = l0[j]; bfrag0[8 + j] = h0[j];
                bfrag1[j] = l1[j]; bfrag1[8 + j] = h1[j];
            }
        } else {
#pragma unroll
            for (int j = 0; j < 16; ++j) {
                bfrag0[j] = (__bf16)bp0[(size_t)(kb + j) * Dd];
                bfrag1[j] = (__bf16)bp1[(size_t)(kb + j) * Dd];
            }
        }
#if defined(USE_ASYNC_LDS)
        __builtin_amdgcn_s_wait_asynccnt(0);
#endif
        __syncthreads();
#pragma unroll
        for (int mf = 0; mf < 4; ++mf) {
            const __bf16* ar = &sA[(mf * 16 + lh) * 32 + hi * 16];
            v8bf lo = *(const v8bf*)(ar);
            v8bf hs = *(const v8bf*)(ar + 8);
            v16bf afrag;
#pragma unroll
            for (int j = 0; j < 8; ++j) { afrag[j] = lo[j]; afrag[8 + j] = hs[j]; }
            acc[mf][0] = __builtin_amdgcn_wmma_f32_16x16x32_bf16(
                false, afrag, false, bfrag0, (short)0, acc[mf][0], false, false);
            acc[mf][1] = __builtin_amdgcn_wmma_f32_16x16x32_bf16(
                false, afrag, false, bfrag1, (short)0, acc[mf][1], false, false);
        }
    }
#pragma unroll
    for (int mf = 0; mf < 4; ++mf) {
#pragma unroll
        for (int i = 0; i < 8; ++i) {
            int m = m0 + mf * 16 + i + hi * 8;
            if (m < cnt) {
                int id = elist[m];
                float* orow = out + (size_t)(id >> 1) * Dd;
#pragma unroll
                for (int nf = 0; nf < 2; ++nf)
                    atomicAdd(orow + nbase + nf * 16 + lh, acc[mf][nf][i]);
            }
        }
    }
}

// ---------------- launch ----------------
extern "C" void kernel_launch(void* const* d_in, const int* in_sizes, int n_in,
                              void* d_out, int out_size, void* d_ws, size_t ws_size,
                              hipStream_t stream) {
    const float* x  = (const float*)d_in[0];
    const float* rw = (const float*)d_in[1];
    const float* w1 = (const float*)d_in[2];
    const float* w2 = (const float*)d_in[3];
    float* out = (float*)d_out;

    char* ws = (char*)d_ws;
    int*    counts = (int*)(ws + WS_COUNTS);
    int*    list   = (int*)(ws + WS_LIST);
    float*  gates  = (float*)(ws + WS_GATES);
    __bf16* hbuf   = (__bf16*)(ws + WS_HBUF);
    __bf16* w1t    = (__bf16*)(ws + WS_W1T);
    __bf16* w2t    = (__bf16*)(ws + WS_W2T);

    const int use_tr = (ws_size >= WS_NEED) ? 1 : 0;

    moe_zero<<<2048, 256, 0, stream>>>(out, Nn * Dd, counts);
    moe_router<<<Nn / 8, 256, 0, stream>>>(x, rw, counts, list, gates);
    if (use_tr) {
        moe_transpose<<<dim3(Hh / 32, Dd / 32, Ee), 256, 0, stream>>>(w1, w1t, Dd, Hh);
        moe_transpose<<<dim3(Dd / 32, Hh / 32, Ee), 256, 0, stream>>>(w2, w2t, Hh, Dd);
    }
    moe_ffn1<<<dim3(CAP / 64, Hh / 128, Ee), 128, 0, stream>>>(
        x, w1, w1t, use_tr, counts, list, gates, hbuf);
    moe_ffn2<<<dim3(CAP / 64, Dd / 128, Ee), 128, 0, stream>>>(
        hbuf, w2, w2t, use_tr, counts, list, out);
}
